// OptimizedAttention_68736656605917
// MI455X (gfx1250) — compile-verified, matrix operand order checked by reading, not run on hardware
//
#include <hip/hip_runtime.h>

#define S_LEN   2048
#define DMODEL  1024
#define DK      64
#define QTILE   16      // q rows per q-subtile
#define NQ      2       // q-subtiles per wave (A-fragment reuse factor)
#define KTILE   64      // keys per mainloop iteration
#define NTILES  (S_LEN / KTILE)
#define WAVES   4
#define BLOCK   (WAVES * 32)
#define LDSP    72      // LDS pitch in halves (16B-aligned rows, staggered banks)

typedef __attribute__((ext_vector_type(16))) _Float16 v16h;
typedef __attribute__((ext_vector_type(8)))  _Float16 v8h;
typedef __attribute__((ext_vector_type(8)))  float    v8f;

static __device__ __forceinline__ v16h cat8(v8h a, v8h b) {
    return __builtin_shufflevector(a, b, 0,1,2,3,4,5,6,7,8,9,10,11,12,13,14,15);
}

// Load a 16x32 WMMA A/B fragment from row-major LDS (pitch LDSP halves).
// Lanes 0-15: elems {0..7}=col0+0..7, {8..15}=col0+16..23;
// lanes 16-31: elems {0..7}=col0+8..15, {8..15}=col0+24..31. Row = row0 + lane%16.
static __device__ __forceinline__ v16h lds_frag(const _Float16* base, int row0,
                                                int col0, int hi, int m) {
    const _Float16* r = base + (row0 + m) * LDSP + col0 + hi * 8;
    v8h lo = *(const v8h*)(r);
    v8h hh = *(const v8h*)(r + 16);
    return cat8(lo, hh);
}

__global__ __launch_bounds__(BLOCK)
void OptimizedAttention_68736656605917_kernel(const float* __restrict__ Q,
                                              const float* __restrict__ K,
                                              const float* __restrict__ V,
                                              float* __restrict__ O) {
    // double-buffered tiles: K row-major f16, V transposed f16
    __shared__ __align__(16) _Float16 Klds [2][KTILE * LDSP];
    __shared__ __align__(16) _Float16 VTlds[2][DK * LDSP];

    const int tid  = threadIdx.x;
    const int wave = tid >> 5;
    const int lane = tid & 31;
    const int hi   = lane >> 4;
    const int m    = lane & 15;

    const int qblk  = blockIdx.x;   // 0..15
    const int head  = blockIdx.y;   // 0..15
    const int batch = blockIdx.z;   // 0..1

    const int   qbase = qblk * (WAVES * NQ * QTILE) + wave * (NQ * QTILE);
    const float scale = 0.125f;     // 1/sqrt(64), folded into Q before f16 cvt

    // --- Q fragments (B-operand: lane = q column, d packed per WMMA layout) ---
    v16h qb[NQ][2];
#pragma unroll
    for (int qq = 0; qq < NQ; ++qq) {
        const float* qptr =
            Q + ((size_t)(batch * S_LEN + qbase + qq * QTILE + m)) * DMODEL + head * DK;
#pragma unroll
        for (int dh = 0; dh < 2; ++dh) {
            const float* p = qptr + dh * 32 + hi * 8;
            v16h f;
#pragma unroll
            for (int i = 0; i < 8; ++i) {
                f[i]     = (_Float16)(p[i]      * scale);
                f[i + 8] = (_Float16)(p[i + 16] * scale);
            }
            qb[qq][dh] = f;
        }
    }

    v8f o[NQ][4];
#pragma unroll
    for (int qq = 0; qq < NQ; ++qq)
#pragma unroll
        for (int n = 0; n < 4; ++n) o[qq][n] = {};
    float m_run[NQ], l_run[NQ];
#pragma unroll
    for (int qq = 0; qq < NQ; ++qq) { m_run[qq] = -3.0e38f; l_run[qq] = 0.0f; }

    const float* kbase_g = K + (size_t)batch * S_LEN * DMODEL + head * DK;
    const float* vbase_g = V + (size_t)batch * S_LEN * DMODEL + head * DK;

    const int tr = tid >> 4;          // 0..7  (tile row within a pass)
    const int tc = (tid & 15) * 4;    // 0..60 (float4 column)

    // ---- prologue: load tile 0 into buffer 0 ----
    {
        float4 k4[8], v4[8];
#pragma unroll
        for (int pass = 0; pass < 8; ++pass) {
            const int row = tr + pass * 8;
            k4[pass] = *(const float4*)(kbase_g + (size_t)row * DMODEL + tc);
            v4[pass] = *(const float4*)(vbase_g + (size_t)row * DMODEL + tc);
        }
#pragma unroll
        for (int pass = 0; pass < 8; ++pass) {
            const int row = tr + pass * 8;
            _Float16* kd = Klds[0] + row * LDSP + tc;
            kd[0] = (_Float16)k4[pass].x; kd[1] = (_Float16)k4[pass].y;
            kd[2] = (_Float16)k4[pass].z; kd[3] = (_Float16)k4[pass].w;
            VTlds[0][(tc + 0) * LDSP + row] = (_Float16)v4[pass].x;
            VTlds[0][(tc + 1) * LDSP + row] = (_Float16)v4[pass].y;
            VTlds[0][(tc + 2) * LDSP + row] = (_Float16)v4[pass].z;
            VTlds[0][(tc + 3) * LDSP + row] = (_Float16)v4[pass].w;
        }
    }
    __syncthreads();

    for (int it = 0; it < NTILES; ++it) {
        const int cur = it & 1;
        const int nxt = cur ^ 1;
        const bool have_next = (it + 1) < NTILES;

        // --- stage next tile's global loads NOW (latency overlaps compute) ---
        float4 k4[8], v4[8];
        if (have_next) {
            const int kt1 = (it + 1) * KTILE;
#pragma unroll
            for (int pass = 0; pass < 8; ++pass) {
                const int row = kt1 + tr + pass * 8;
                k4[pass] = *(const float4*)(kbase_g + (size_t)row * DMODEL + tc);
                v4[pass] = *(const float4*)(vbase_g + (size_t)row * DMODEL + tc);
            }
            if (it + 2 < NTILES) {  // prefetch tile i+2 toward L2 (global_prefetch_b8)
                const int kt2 = (it + 2) * KTILE;
                __builtin_prefetch(kbase_g + (size_t)(kt2 + (tid >> 1)) * DMODEL + (tid & 1) * 32, 0, 1);
                __builtin_prefetch(vbase_g + (size_t)(kt2 + (tid >> 1)) * DMODEL + (tid & 1) * 32, 0, 1);
            }
        }

        // --- S^T = K x Q : 4 tiles of 16(key)x16(q) per q-subtile.
        //     Each K-fragment (A) is loaded once and reused for NQ WMMAs. ---
        v8f st[NQ][4];
#pragma unroll
        for (int mt = 0; mt < 4; ++mt) {
            const v16h a0 = lds_frag(Klds[cur], mt * 16, 0, hi, m);
            const v16h a1 = lds_frag(Klds[cur], mt * 16, 32, hi, m);
#pragma unroll
            for (int qq = 0; qq < NQ; ++qq) {
                v8f c = {};
                c = __builtin_amdgcn_wmma_f32_16x16x32_f16(false, a0, false, qb[qq][0], (short)0, c, false, false);
                c = __builtin_amdgcn_wmma_f32_16x16x32_f16(false, a1, false, qb[qq][1], (short)0, c, false, false);
                st[qq][mt] = c;
            }
        }

        // --- online softmax + P fragments (per q-subtile) ---
        v16h pb[NQ][2];
#pragma unroll
        for (int qq = 0; qq < NQ; ++qq) {
            float mx = st[qq][0][0];
#pragma unroll
            for (int mt = 0; mt < 4; ++mt)
#pragma unroll
                for (int r = 0; r < 8; ++r) mx = fmaxf(mx, st[qq][mt][r]);
            mx = fmaxf(mx, __shfl_xor(mx, 16, 32));

            const float m_new = fmaxf(m_run[qq], mx);
            const float alpha = __expf(m_run[qq] - m_new);

            float lsum = 0.0f;
#pragma unroll
            for (int mt = 0; mt < 4; ++mt)
#pragma unroll
                for (int r = 0; r < 8; ++r) {
                    const float e = __expf(st[qq][mt][r] - m_new);
                    st[qq][mt][r] = e;
                    lsum += e;
                }
            lsum += __shfl_xor(lsum, 16, 32);
            l_run[qq] = l_run[qq] * alpha + lsum;
            m_run[qq] = m_new;

#pragma unroll
            for (int n = 0; n < 4; ++n) o[qq][n] *= alpha;

            // accumulator layout of S^T IS the B-operand layout for P
#pragma unroll
            for (int j = 0; j < 2; ++j) {
                v16h f;
#pragma unroll
                for (int i = 0; i < 8; ++i) {
                    f[i]     = (_Float16)st[qq][2 * j][i];
                    f[i + 8] = (_Float16)st[qq][2 * j + 1][i];
                }
                pb[qq][j] = f;
            }
        }

        // --- O^T += V^T x P : each V^T fragment reused for NQ WMMAs ---
#pragma unroll
        for (int n = 0; n < 4; ++n) {
            const v16h a0 = lds_frag(VTlds[cur], n * 16, 0, hi, m);
            const v16h a1 = lds_frag(VTlds[cur], n * 16, 32, hi, m);
#pragma unroll
            for (int qq = 0; qq < NQ; ++qq) {
                o[qq][n] = __builtin_amdgcn_wmma_f32_16x16x32_f16(false, a0, false, pb[qq][0], (short)0, o[qq][n], false, false);
                o[qq][n] = __builtin_amdgcn_wmma_f32_16x16x32_f16(false, a1, false, pb[qq][1], (short)0, o[qq][n], false, false);
            }
        }

        // --- drain staged loads into the other buffer (after compute) ---
        if (have_next) {
#pragma unroll
            for (int pass = 0; pass < 8; ++pass) {
                const int row = tr + pass * 8;
                _Float16* kd = Klds[nxt] + row * LDSP + tc;
                kd[0] = (_Float16)k4[pass].x; kd[1] = (_Float16)k4[pass].y;
                kd[2] = (_Float16)k4[pass].z; kd[3] = (_Float16)k4[pass].w;
                VTlds[nxt][(tc + 0) * LDSP + row] = (_Float16)v4[pass].x;
                VTlds[nxt][(tc + 1) * LDSP + row] = (_Float16)v4[pass].y;
                VTlds[nxt][(tc + 2) * LDSP + row] = (_Float16)v4[pass].z;
                VTlds[nxt][(tc + 3) * LDSP + row] = (_Float16)v4[pass].w;
            }
        }
        __syncthreads();
    }

    // --- normalize and store: lane holds q=m, d = n*16 + hi*8 + r (contig 8) ---
#pragma unroll
    for (int qq = 0; qq < NQ; ++qq) {
        const float inv = 1.0f / l_run[qq];
        float* obase = O + ((size_t)(batch * S_LEN + qbase + qq * QTILE + m)) * DMODEL
                         + head * DK + hi * 8;
#pragma unroll
        for (int n = 0; n < 4; ++n) {
            float4 r0, r1;
            r0.x = o[qq][n][0] * inv; r0.y = o[qq][n][1] * inv;
            r0.z = o[qq][n][2] * inv; r0.w = o[qq][n][3] * inv;
            r1.x = o[qq][n][4] * inv; r1.y = o[qq][n][5] * inv;
            r1.z = o[qq][n][6] * inv; r1.w = o[qq][n][7] * inv;
            *(float4*)(obase + n * 16)     = r0;
            *(float4*)(obase + n * 16 + 4) = r1;
        }
    }
}

extern "C" void kernel_launch(void* const* d_in, const int* in_sizes, int n_in,
                              void* d_out, int out_size, void* d_ws, size_t ws_size,
                              hipStream_t stream) {
    (void)in_sizes; (void)n_in; (void)out_size; (void)d_ws; (void)ws_size;
    const float* Q = (const float*)d_in[0];
    const float* K = (const float*)d_in[1];
    const float* V = (const float*)d_in[2];
    float* O = (float*)d_out;

    dim3 grid(S_LEN / (WAVES * NQ * QTILE), 16 /*heads*/, 2 /*batch*/);
    dim3 block(BLOCK);
    OptimizedAttention_68736656605917_kernel<<<grid, block, 0, stream>>>(Q, K, V, O);
}
